// TemporalGCNLinkPredictor_57123065037361
// MI455X (gfx1250) — matrix operand order repaired and checked by reading, not compiled, not run on hardware
//
#include <hip/hip_runtime.h>
#include <hip/hip_bf16.h>
#include <math.h>

// ---------------------------------------------------------------------------
// TemporalGCN on gfx1250: WMMA bf16 GEMMs (LDS-staged weights) + atomic scatter.
// T=8, N=50000, D=H=128, E=800000
// ---------------------------------------------------------------------------

#define H_DIM 128

typedef __attribute__((ext_vector_type(16))) __bf16 v16bf;
typedef __attribute__((ext_vector_type(8)))  float  v8f;

__device__ __forceinline__ float sigmoidf_(float x) {
    return 1.0f / (1.0f + __expf(-x));
}

__device__ __forceinline__ void atomicAddF32(float* p, float v) {
    (void)__hip_atomic_fetch_add(p, v, __ATOMIC_RELAXED, __HIP_MEMORY_SCOPE_AGENT);
}

// ---------------------------------------------------------------------------
// WMMA GEMM:  C[N x 128] = A[N x K] * Wt + bias
//   Wt is bf16, stored column-major: Wt[col * K + k], K = KT*32.
//   KT==4 : A = A0 (N x 128)
//   KT==8 : A = concat(A0, A1) along K (each N x 128) -- gate GEMMs.
// Block = 256 threads = 8 waves; each wave computes a 16-row x 128-col strip,
// so a block covers 128 rows. The whole weight matrix is staged in LDS once
// per block; B fragments then come from ds_load instead of global memory.
// ---------------------------------------------------------------------------
template <int KT>
__global__ __launch_bounds__(256)
void gemm_wmma_bf16(const float* __restrict__ A0, const float* __restrict__ A1,
                    const __bf16* __restrict__ Wt, const float* __restrict__ bias,
                    float* __restrict__ C, int nRows)
{
    constexpr int Kw = KT * 32;
    __shared__ __align__(32) __bf16 sW[Kw * H_DIM];   // 32KB (KT=4) / 64KB (KT=8)

    // ---- cooperative weight stage: global -> LDS (16B per thread per iter) --
    {
        const uint4* gsrc = (const uint4*)Wt;          // Kw*128 bf16 = KT*512 uint4
        uint4* sdst = (uint4*)sW;
        #pragma unroll
        for (int i = 0; i < KT * 2; ++i)
            sdst[threadIdx.x + i * 256] = gsrc[threadIdx.x + i * 256];
    }
    __syncthreads();

    const int lane = threadIdx.x & 31;
    const int wave = threadIdx.x >> 5;
    const int half = lane >> 4;      // 0: lanes 0-15, 1: lanes 16-31
    const int l15  = lane & 15;
    const int rowBase = blockIdx.x * 128 + wave * 16;

    int aRow = rowBase + l15;
    if (aRow >= nRows) aRow = nRows - 1;   // clamp (stores are guarded)

    v8f acc[8];
    #pragma unroll
    for (int t = 0; t < 8; ++t) {
        #pragma unroll
        for (int e = 0; e < 8; ++e) acc[t][e] = 0.0f;
    }

    #pragma unroll
    for (int kb = 0; kb < KT; ++kb) {
        const float* Asrc = (KT == 8 && kb >= 4) ? A1 : A0;
        const int kLocal = (kb & 3) * 32;
        const float* ap = Asrc + (size_t)aRow * H_DIM;

        // A fragment (16x32, MxK) in this code's contiguous k order: lane half selects K sub-ranges.
        // elements 0..7  -> K = kLocal + half*16 + j
        // elements 8..15 -> K = kLocal + half*16 + 8 + j
        const int k0 = kLocal + half * 16;
        const int k1 = kLocal + half * 16 + 8;
        float4 a0 = *(const float4*)(ap + k0);
        float4 a1 = *(const float4*)(ap + k0 + 4);
        float4 a2 = *(const float4*)(ap + k1);
        float4 a3 = *(const float4*)(ap + k1 + 4);

        v16bf af;
        af[0]  = (__bf16)a0.x; af[1]  = (__bf16)a0.y; af[2]  = (__bf16)a0.z; af[3]  = (__bf16)a0.w;
        af[4]  = (__bf16)a1.x; af[5]  = (__bf16)a1.y; af[6]  = (__bf16)a1.z; af[7]  = (__bf16)a1.w;
        af[8]  = (__bf16)a2.x; af[9]  = (__bf16)a2.y; af[10] = (__bf16)a2.z; af[11] = (__bf16)a2.w;
        af[12] = (__bf16)a3.x; af[13] = (__bf16)a3.y; af[14] = (__bf16)a3.z; af[15] = (__bf16)a3.w;

        // B fragment (32x16, KxN), same k order as A: lane l15 = column, half selects K 0-15 / 16-31.
        // Column-major bf16 weights in LDS -> contiguous 32B ds read per lane.
        const int kw = kb * 32 + half * 16;
        #pragma unroll
        for (int t = 0; t < 8; ++t) {
            const int col = t * 16 + l15;
            v16bf bf = *(const v16bf*)(sW + col * Kw + kw);
            acc[t] = __builtin_amdgcn_wmma_f32_16x16x32_bf16(
                false, af, false, bf, (short)0, acc[t], false, false);
        }
    }

    // Epilogue: C/D layout -- VGPR v, lane l: M = v + 8*half, N(col) = l15.
    #pragma unroll
    for (int t = 0; t < 8; ++t) {
        const int col = t * 16 + l15;
        const float b = bias ? bias[col] : 0.0f;
        #pragma unroll
        for (int v = 0; v < 8; ++v) {
            const int r = rowBase + v + 8 * half;
            if (r < nRows) C[(size_t)r * H_DIM + col] = acc[t][v] + b;
        }
    }
}

// ---------------------------------------------------------------------------
// Small helper kernels
// ---------------------------------------------------------------------------
__global__ void k_convert_weight(const float* __restrict__ W, __bf16* __restrict__ Wt, int K)
{
    int idx = blockIdx.x * blockDim.x + threadIdx.x;
    if (idx >= K * H_DIM) return;
    int k = idx / H_DIM;
    int c = idx - k * H_DIM;
    Wt[(size_t)c * K + k] = (__bf16)W[idx];     // row-major -> column-major bf16
}

__global__ void k_fill(float* __restrict__ p, float v, int n)
{
    int i = blockIdx.x * blockDim.x + threadIdx.x;
    if (i < n) p[i] = v;
}

__global__ void k_count_deg(const int* __restrict__ dst, float* __restrict__ deg, int e)
{
    int i = blockIdx.x * blockDim.x + threadIdx.x;
    if (i < e) atomicAddF32(&deg[dst[i]], 1.0f);
}

__global__ void k_rsqrt(float* __restrict__ p, int n)
{
    int i = blockIdx.x * blockDim.x + threadIdx.x;
    if (i < n) p[i] = rsqrtf(p[i]);
}

// agg = lin * dinv^2 + bias    (self-loop term + bias of the GCN conv)
__global__ void k_self_init(const float* __restrict__ lin, const float* __restrict__ dinv,
                            const float* __restrict__ bias, float* __restrict__ agg, int nN)
{
    int i = blockIdx.x * blockDim.x + threadIdx.x;
    if (i >= nN * H_DIM) return;
    int node = i >> 7;
    int c = i & (H_DIM - 1);
    float di = dinv[node];
    agg[i] = lin[i] * di * di + bias[c];
}

// agg[dst] += lin[src] * dinv[src]*dinv[dst]   (one wave per edge, 4 floats/lane)
__global__ __launch_bounds__(256)
void k_scatter(const float* __restrict__ lin, const float* __restrict__ dinv,
               const int* __restrict__ src, const int* __restrict__ dst,
               float* __restrict__ agg, int nE)
{
    int e = blockIdx.x * 8 + (threadIdx.x >> 5);
    if (e >= nE) return;
    int lane = threadIdx.x & 31;
    int s = src[e];
    int d = dst[e];
    float coef = dinv[s] * dinv[d];
    const float* hp = lin + (size_t)s * H_DIM;
    float* ap = agg + (size_t)d * H_DIM;
    float4 hv = *(const float4*)(hp + lane * 4);
    atomicAddF32(ap + lane * 4 + 0, hv.x * coef);
    atomicAddF32(ap + lane * 4 + 1, hv.y * coef);
    atomicAddF32(ap + lane * 4 + 2, hv.z * coef);
    atomicAddF32(ap + lane * 4 + 3, hv.w * coef);
}

__global__ void k_relu(float* __restrict__ x, int n)
{
    int i = blockIdx.x * blockDim.x + threadIdx.x;
    if (i < n) x[i] = fmaxf(x[i], 0.0f);
}

// rlin <- sigmoid(rlin) * h   (produces r * h_prev in place)
__global__ void k_rh(float* __restrict__ rlin, const float* __restrict__ h, int n)
{
    int i = blockIdx.x * blockDim.x + threadIdx.x;
    if (i < n) {
        float r = sigmoidf_(rlin[i]);
        rlin[i] = r * h[i];
    }
}

// h <- (1-u)*h + u*tanh(clin), u = sigmoid(ulin)
__global__ void k_gru(const float* __restrict__ ulin, const float* __restrict__ clin,
                      float* __restrict__ h, int n)
{
    int i = blockIdx.x * blockDim.x + threadIdx.x;
    if (i < n) {
        float u = sigmoidf_(ulin[i]);
        float c = tanhf(clin[i]);
        h[i] = (1.0f - u) * h[i] + u * c;
    }
}

// ---------------------------------------------------------------------------
// Host driver
// ---------------------------------------------------------------------------
extern "C" void kernel_launch(void* const* d_in, const int* in_sizes, int n_in,
                              void* d_out, int out_size, void* d_ws, size_t ws_size,
                              hipStream_t stream)
{
    const int T = 8, N = 50000, E = 800000;
    const int NH = N * H_DIM;

    const float* x_seq = (const float*)d_in[0];
    const int*   ei    = (const int*)d_in[1];

    const float* W_in0 = (const float*)d_in[2];
    const float* b_in0 = (const float*)d_in[3];
    const float* W_h0  = (const float*)d_in[4];
    const float* b_h0  = (const float*)d_in[5];
    const float* Wu0   = (const float*)d_in[6];
    const float* bu0   = (const float*)d_in[7];
    const float* Wr0   = (const float*)d_in[8];
    const float* br0   = (const float*)d_in[9];
    const float* Wc0   = (const float*)d_in[10];
    const float* bc0   = (const float*)d_in[11];
    const float* W_in1 = (const float*)d_in[12];
    const float* b_in1 = (const float*)d_in[13];
    const float* W_h1  = (const float*)d_in[14];
    const float* b_h1  = (const float*)d_in[15];
    const float* Wu1   = (const float*)d_in[16];
    const float* bu1   = (const float*)d_in[17];
    const float* Wr1   = (const float*)d_in[18];
    const float* br1   = (const float*)d_in[19];
    const float* Wc1   = (const float*)d_in[20];
    const float* bc1   = (const float*)d_in[21];
    const float* W_out = (const float*)d_in[22];
    const float* b_out = (const float*)d_in[23];

    float* out = (float*)d_out;

    // ---- workspace carve-up -------------------------------------------------
    char* ws = (char*)d_ws;
    size_t off = 0;
    auto carve = [&](size_t bytes) -> char* {
        char* p = ws + off;
        off = (off + bytes + 255) & ~(size_t)255;
        return p;
    };
    float* dinv = (float*)carve((size_t)N * 4);
    float* lin  = (float*)carve((size_t)NH * 4);   // GEMM scratch / ULIN
    float* cur  = (float*)carve((size_t)NH * 4);   // GCN output
    float* rbuf = (float*)carve((size_t)NH * 4);   // RLIN -> r*h
    float* cbuf = (float*)carve((size_t)NH * 4);   // CLIN
    float* h0   = (float*)carve((size_t)NH * 4);
    float* h1   = (float*)carve((size_t)NH * 4);

    __bf16* Wt_in0 = (__bf16*)carve((size_t)128 * 128 * 2);
    __bf16* Wt_h0  = (__bf16*)carve((size_t)128 * 128 * 2);
    __bf16* Wt_u0  = (__bf16*)carve((size_t)256 * 128 * 2);
    __bf16* Wt_r0  = (__bf16*)carve((size_t)256 * 128 * 2);
    __bf16* Wt_c0  = (__bf16*)carve((size_t)256 * 128 * 2);
    __bf16* Wt_in1 = (__bf16*)carve((size_t)128 * 128 * 2);
    __bf16* Wt_h1  = (__bf16*)carve((size_t)128 * 128 * 2);
    __bf16* Wt_u1  = (__bf16*)carve((size_t)256 * 128 * 2);
    __bf16* Wt_r1  = (__bf16*)carve((size_t)256 * 128 * 2);
    __bf16* Wt_c1  = (__bf16*)carve((size_t)256 * 128 * 2);
    __bf16* Wt_out = (__bf16*)carve((size_t)128 * 128 * 2);

    const int BT = 256;
    auto gridFor = [&](int n) { return dim3((n + BT - 1) / BT); };
    const dim3 gridGemm((N + 127) / 128);
    const dim3 blkGemm(256);

    // ---- weight conversion (fp32 row-major -> bf16 column-major) -----------
    auto convW = [&](const float* W, __bf16* Wt, int K) {
        k_convert_weight<<<gridFor(K * H_DIM), BT, 0, stream>>>(W, Wt, K);
    };
    convW(W_in0, Wt_in0, 128); convW(W_h0, Wt_h0, 128);
    convW(Wu0,   Wt_u0,  256); convW(Wr0,  Wt_r0, 256); convW(Wc0, Wt_c0, 256);
    convW(W_in1, Wt_in1, 128); convW(W_h1, Wt_h1, 128);
    convW(Wu1,   Wt_u1,  256); convW(Wr1,  Wt_r1, 256); convW(Wc1, Wt_c1, 256);
    convW(W_out, Wt_out, 128);

    // ---- init hidden states -------------------------------------------------
    k_fill<<<gridFor(NH), BT, 0, stream>>>(h0, 0.0f, NH);
    k_fill<<<gridFor(NH), BT, 0, stream>>>(h1, 0.0f, NH);

    // ---- GCN conv: out = D^-1/2 (A+I) D^-1/2 (X @ W) + b --------------------
    auto gcn = [&](const float* X, const __bf16* WT, const float* b,
                   const int* src, const int* dst, float* outBuf) {
        gemm_wmma_bf16<4><<<gridGemm, blkGemm, 0, stream>>>(X, nullptr, WT, nullptr, lin, N);
        k_self_init<<<gridFor(NH), BT, 0, stream>>>(lin, dinv, b, outBuf, N);
        k_scatter<<<dim3((E + 7) / 8), 256, 0, stream>>>(lin, dinv, src, dst, outBuf, E);
    };

    // ---- one GCN-GRU cell ---------------------------------------------------
    auto cell = [&](const float* X, float* h,
                    const __bf16* WTin, const float* bin,
                    const __bf16* WTh,  const float* bh,
                    const __bf16* WTu,  const float* bu,
                    const __bf16* WTr,  const float* br,
                    const __bf16* WTc,  const float* bc,
                    const int* src, const int* dst) {
        gcn(X, WTin, bin, src, dst, cur);
        k_relu<<<gridFor(NH), BT, 0, stream>>>(cur, NH);
        gcn(cur, WTh, bh, src, dst, cur);   // stream-serialized: safe in-place
        // gates: comb = [cur, h]  (never materialized)
        gemm_wmma_bf16<8><<<gridGemm, blkGemm, 0, stream>>>(cur, h, WTu, bu, lin, N);   // ULIN
        gemm_wmma_bf16<8><<<gridGemm, blkGemm, 0, stream>>>(cur, h, WTr, br, rbuf, N);  // RLIN
        k_rh<<<gridFor(NH), BT, 0, stream>>>(rbuf, h, NH);                              // r*h
        gemm_wmma_bf16<8><<<gridGemm, blkGemm, 0, stream>>>(cur, rbuf, WTc, bc, cbuf, N); // CLIN
        k_gru<<<gridFor(NH), BT, 0, stream>>>(lin, cbuf, h, NH);
    };

    // ---- time loop ----------------------------------------------------------
    for (int t = 0; t < T; ++t) {
        const int* src = ei + (size_t)t * 2 * E;
        const int* dst = src + E;
        const float* x_t = x_seq + (size_t)t * NH;

        // degree -> dinv = rsqrt(deg + 1)
        k_fill<<<gridFor(N), BT, 0, stream>>>(dinv, 1.0f, N);
        k_count_deg<<<gridFor(E), BT, 0, stream>>>(dst, dinv, E);
        k_rsqrt<<<gridFor(N), BT, 0, stream>>>(dinv, N);

        cell(x_t, h0, Wt_in0, b_in0, Wt_h0, b_h0, Wt_u0, bu0, Wt_r0, br0, Wt_c0, bc0, src, dst);
        cell(h0,  h1, Wt_in1, b_in1, Wt_h1, b_h1, Wt_u1, bu1, Wt_r1, br1, Wt_c1, bc1, src, dst);

        // emb = h1 @ W_out + b_out
        gemm_wmma_bf16<4><<<gridGemm, blkGemm, 0, stream>>>(
            h1, nullptr, Wt_out, b_out, out + (size_t)t * NH, N);
    }
}
